// DGCNN_50594714747409
// MI455X (gfx1250) — matrix operand order repaired and checked by reading, not compiled, not run on hardware
//
#include <hip/hip_runtime.h>
#include <hip/hip_bf16.h>

typedef _Float16 f16;
typedef __attribute__((ext_vector_type(8)))  _Float16 v8h;
typedef __attribute__((ext_vector_type(16))) _Float16 v16h;
typedef __attribute__((ext_vector_type(8)))  float    v8f;

#define NPTS 1024
#define BATCH 16
#define KNN 20
#define EPSBN 1e-5f

// ---------------- prep kernels ----------------

// x [16][3][1024] f32 -> x0pad [16][1024][32] f16 (channels 3..31 = 0)
__global__ void prep_x0_k(const float* __restrict__ x, f16* __restrict__ o) {
    int i = blockIdx.x * blockDim.x + threadIdx.x;         // 16*1024*32
    if (i >= BATCH * NPTS * 32) return;
    int c = i & 31, n = (i >> 5) & 1023, b = i >> 15;
    float v = (c < 3) ? x[(b * 3 + c) * NPTS + n] : 0.f;
    o[i] = (f16)v;
}

// w [O][2C] f32 -> W' [2O][Kpad] f16: rows 0..O-1 = W_a, rows O..2O-1 = W_b - W_a
__global__ void prep_edge_w_k(const float* __restrict__ w, int O, int C, int Kpad,
                              f16* __restrict__ o) {
    int i = blockIdx.x * blockDim.x + threadIdx.x;
    if (i >= 2 * O * Kpad) return;
    int c = i % Kpad, r = i / Kpad;
    float v = 0.f;
    if (c < C) {
        if (r < O) v = w[r * 2 * C + c];
        else       v = w[(r - O) * 2 * C + C + c] - w[(r - O) * 2 * C + c];
    }
    o[i] = (f16)v;
}

// w [O][C] f32 -> [Opad][C] f16 (zero pad extra rows)
__global__ void prep_w_k(const float* __restrict__ w, int O, int C, int Opad,
                         f16* __restrict__ o) {
    int i = blockIdx.x * blockDim.x + threadIdx.x;
    if (i >= Opad * C) return;
    int r = i / C;
    o[i] = (r < O) ? (f16)w[i] : (f16)0.f;
}

// ---------------- squared norms ----------------
__global__ void sqsum_k(const f16* __restrict__ xt, int ldx, int coff, int C,
                        float* __restrict__ xx) {
    int i = blockIdx.x * blockDim.x + threadIdx.x;         // b*1024+n
    if (i >= BATCH * NPTS) return;
    const f16* p = xt + (size_t)i * ldx + coff;
    float s = 0.f;
    for (int c = 0; c < C; c++) { float f = (float)p[c]; s += f * f; }
    xx[i] = s;
}

// ---------------- fused pairwise-dist + top-k (one wave per point) ----------------
__global__ void topk_k(const f16* __restrict__ xt, int ldx, int coff, int C,
                       const float* __restrict__ xx, int* __restrict__ idx) {
    int wid = (blockIdx.x * blockDim.x + threadIdx.x) >> 5;  // row id (wave-uniform)
    int lane = threadIdx.x & 31;
    if (wid >= BATCH * NPTS) return;
    int b = wid >> 10, n = wid & 1023;
    const f16* base = xt + ((size_t)b * NPTS) * ldx + coff;
    const f16* xn = base + (size_t)n * ldx;

    float v[32];
#pragma unroll
    for (int t = 0; t < 32; t++) v[t] = 0.f;
    for (int c = 0; c < C; c++) {
        float xnc = (float)xn[c];
#pragma unroll
        for (int t = 0; t < 32; t++)
            v[t] += xnc * (float)base[(size_t)(t * 32 + lane) * ldx + c];
    }
    float xxn = xx[wid];
#pragma unroll
    for (int t = 0; t < 32; t++) {
        int m = t * 32 + lane;
        v[t] = 2.f * v[t] - xxn - xx[(b << 10) + m];
    }
    // 20 iterative argmax passes, tie-break on smaller index (matches top_k order)
    for (int p = 0; p < KNN; p++) {
        float bv = -3.0e38f; int bm = 0x7fffffff;
#pragma unroll
        for (int t = 0; t < 32; t++) {
            int m = t * 32 + lane;
            if (v[t] > bv || (v[t] == bv && m < bm)) { bv = v[t]; bm = m; }
        }
        for (int off = 16; off; off >>= 1) {
            float ov = __shfl_xor(bv, off, 32);
            int   om = __shfl_xor(bm, off, 32);
            if (ov > bv || (ov == bv && om < bm)) { bv = ov; bm = om; }
        }
        if (lane == 0) idx[wid * KNN + p] = bm;
        if ((bm & 31) == lane) {
#pragma unroll
            for (int t = 0; t < 32; t++)
                if (t == (bm >> 5)) v[t] = -3.0e38f;
        }
    }
}

// ---------------- generic WMMA GEMM: Z[b] = A(M x K) @ X[b](K x N), N=1024 ----------------
// A f16 [Mtiles*16][lda] row-major; X f16 [b][1024][ldx], B-column n = X[b][n][coff..]
__global__ void gemm_wmma_k(const f16* __restrict__ A, int lda, int Mtiles,
                            const f16* __restrict__ X, int ldx, int coff, int K,
                            float* __restrict__ Z) {
    int tid = (blockIdx.x * blockDim.x + threadIdx.x) >> 5;
    int perB = Mtiles * 64;
    if (tid >= perB * BATCH) return;
    int b = tid / perB, r = tid - b * perB;
    int mt = r >> 6, nt = r & 63;
    int lane = threadIdx.x & 31, half = lane >> 4, l16 = lane & 15;

    const f16* arow = A + (size_t)(mt * 16 + l16) * lda;
    const f16* bcol = X + ((size_t)b * NPTS + nt * 16 + l16) * ldx + coff + half * 16;
    v8f acc = {};
    for (int k0 = 0; k0 < K; k0 += 32) {
        v8h al = *(const v8h*)(arow + k0 + half * 16);
        v8h ah = *(const v8h*)(arow + k0 + half * 16 + 8);
        v16h a = __builtin_shufflevector(al, ah, 0,1,2,3,4,5,6,7,8,9,10,11,12,13,14,15);
        v8h bl = *(const v8h*)(bcol + k0);
        v8h bh = *(const v8h*)(bcol + k0 + 8);
        v16h bmat = __builtin_shufflevector(bl, bh, 0,1,2,3,4,5,6,7,8,9,10,11,12,13,14,15);
        acc = __builtin_amdgcn_wmma_f32_16x16x32_f16(false, a, false, bmat,
                                                     (short)0, acc, false, false);
    }
    float* zb = Z + (size_t)b * (Mtiles * 16) * NPTS;
    int n = nt * 16 + l16;
#pragma unroll
    for (int j = 0; j < 8; j++)
        zb[(size_t)(mt * 16 + half * 8 + j) * NPTS + n] = acc[j];
}

// ---------------- EdgeConv epilogue: out = act(Q + max_k P[idx]) -> xcat slice ----------------
__global__ void edge_max_k(const float* __restrict__ Z, int O, const int* __restrict__ idx,
                           const float* __restrict__ g, const float* __restrict__ bb,
                           const float* __restrict__ mm, const float* __restrict__ vv,
                           f16* __restrict__ xcat, int ldout, int coffOut) {
    int i = blockIdx.x * blockDim.x + threadIdx.x;   // b*O*1024
    if (i >= BATCH * O * NPTS) return;
    int n = i & 1023, o = (i >> 10) % O, b = i / (O * NPTS);
    const float* zb = Z + (size_t)b * 2 * O * NPTS;
    const float* prow = zb + (size_t)o * NPTS;
    float q = zb[(size_t)(O + o) * NPTS + n];
    const int* id = idx + ((size_t)(b << 10) + n) * KNN;
    float mx = -3.0e38f, mn = 3.0e38f;
#pragma unroll
    for (int p = 0; p < KNN; p++) {
        float v = prow[id[p]];
        mx = fmaxf(mx, v); mn = fminf(mn, v);
    }
    float s = g[o] * rsqrtf(vv[o] + EPSBN);
    float z = q + (s >= 0.f ? mx : mn);
    float y = (z - mm[o]) * s + bb[o];
    y = (y >= 0.f) ? y : 0.2f * y;
    xcat[((size_t)b * NPTS + n) * ldout + coffOut + o] = (f16)y;
}

// ---------------- conv5 GEMM with fused BN+LReLU + pooled partials ----------------
__global__ void gemm5_wmma_k(const f16* __restrict__ A, const f16* __restrict__ X,
                             const float* __restrict__ g, const float* __restrict__ bb,
                             const float* __restrict__ mm, const float* __restrict__ vv,
                             float* __restrict__ pmax, float* __restrict__ psum) {
    const int Mtiles = 64, K = 512, ldx = 512;
    int tid = (blockIdx.x * blockDim.x + threadIdx.x) >> 5;
    int perB = Mtiles * 64;
    if (tid >= perB * BATCH) return;
    int b = tid / perB, r = tid - b * perB;
    int mt = r >> 6, nt = r & 63;
    int lane = threadIdx.x & 31, half = lane >> 4, l16 = lane & 15;

    const f16* arow = A + (size_t)(mt * 16 + l16) * K;
    const f16* bcol = X + ((size_t)b * NPTS + nt * 16 + l16) * ldx + half * 16;
    v8f acc = {};
    for (int k0 = 0; k0 < K; k0 += 32) {
        v8h al = *(const v8h*)(arow + k0 + half * 16);
        v8h ah = *(const v8h*)(arow + k0 + half * 16 + 8);
        v16h a = __builtin_shufflevector(al, ah, 0,1,2,3,4,5,6,7,8,9,10,11,12,13,14,15);
        v8h bl = *(const v8h*)(bcol + k0);
        v8h bh = *(const v8h*)(bcol + k0 + 8);
        v16h bmat = __builtin_shufflevector(bl, bh, 0,1,2,3,4,5,6,7,8,9,10,11,12,13,14,15);
        acc = __builtin_amdgcn_wmma_f32_16x16x32_f16(false, a, false, bmat,
                                                     (short)0, acc, false, false);
    }
#pragma unroll
    for (int j = 0; j < 8; j++) {
        int o = mt * 16 + half * 8 + j;
        float s = g[o] * rsqrtf(vv[o] + EPSBN);
        float y = (acc[j] - mm[o]) * s + bb[o];
        y = (y >= 0.f) ? y : 0.2f * y;
        float mx = y, sm = y;
        for (int off = 1; off < 16; off <<= 1) {
            mx = fmaxf(mx, __shfl_xor(mx, off, 32));
            sm += __shfl_xor(sm, off, 32);
        }
        if (l16 == 0) {
            size_t pi = ((size_t)(b << 10) + o) * 64 + nt;
            pmax[pi] = mx; psum[pi] = sm;
        }
    }
}

// deterministic tree-free final pool: p16 = [max | mean], f16 [16][2048]
__global__ void pool_reduce_k(const float* __restrict__ pmax, const float* __restrict__ psum,
                              f16* __restrict__ p16) {
    int i = blockIdx.x * blockDim.x + threadIdx.x;   // b*1024+o
    if (i >= BATCH * NPTS) return;
    float mx = -3.0e38f, sm = 0.f;
    for (int t = 0; t < 64; t++) { mx = fmaxf(mx, pmax[(size_t)i * 64 + t]); sm += psum[(size_t)i * 64 + t]; }
    int b = i >> 10, o = i & 1023;
    p16[b * 2048 + o]        = (f16)mx;
    p16[b * 2048 + 1024 + o] = (f16)(sm * (1.f / 1024.f));
}

// ---------------- FC layers: single M-tile (batch=16) WMMA GEMM + epilogue ----------------
// mode 0: BN+lrelu -> outH ; mode 1: +bias, BN+lrelu -> outH ; mode 2: +bias -> outF (col<101)
__global__ void fc_wmma_k(const f16* __restrict__ A, const f16* __restrict__ W, int K,
                          int mode, const float* __restrict__ g, const float* __restrict__ bb,
                          const float* __restrict__ mm, const float* __restrict__ vv,
                          const float* __restrict__ bias,
                          f16* __restrict__ outH, float* __restrict__ outF, int Nout) {
    int nt = blockIdx.x;
    int lane = threadIdx.x & 31, half = lane >> 4, l16 = lane & 15;
    const f16* arow = A + (size_t)l16 * K;
    const f16* bcol = W + (size_t)(nt * 16 + l16) * K + half * 16;
    v8f acc = {};
    for (int k0 = 0; k0 < K; k0 += 32) {
        v8h al = *(const v8h*)(arow + k0 + half * 16);
        v8h ah = *(const v8h*)(arow + k0 + half * 16 + 8);
        v16h a = __builtin_shufflevector(al, ah, 0,1,2,3,4,5,6,7,8,9,10,11,12,13,14,15);
        v8h bl = *(const v8h*)(bcol + k0);
        v8h bh = *(const v8h*)(bcol + k0 + 8);
        v16h bmat = __builtin_shufflevector(bl, bh, 0,1,2,3,4,5,6,7,8,9,10,11,12,13,14,15);
        acc = __builtin_amdgcn_wmma_f32_16x16x32_f16(false, a, false, bmat,
                                                     (short)0, acc, false, false);
    }
    int col = nt * 16 + l16;
#pragma unroll
    for (int j = 0; j < 8; j++) {
        int m = half * 8 + j;          // batch row
        float y = acc[j];
        if (mode == 2) {
            if (col < 101) outF[m * 101 + col] = y + bias[col];
        } else {
            if (mode == 1) y += bias[col];
            float s = g[col] * rsqrtf(vv[col] + EPSBN);
            y = (y - mm[col]) * s + bb[col];
            y = (y >= 0.f) ? y : 0.2f * y;
            outH[m * Nout + col] = (f16)y;
        }
    }
}

// ---------------- host ----------------
extern "C" void kernel_launch(void* const* d_in, const int* in_sizes, int n_in,
                              void* d_out, int out_size, void* d_ws, size_t ws_size,
                              hipStream_t stream) {
    (void)in_sizes; (void)n_in; (void)out_size; (void)ws_size;
    const float* x   = (const float*)d_in[0];
    const float* w1  = (const float*)d_in[1];
    const float* w2  = (const float*)d_in[2];
    const float* w3  = (const float*)d_in[3];
    const float* w4  = (const float*)d_in[4];
    const float* w5  = (const float*)d_in[5];
    const float* wl1 = (const float*)d_in[6];
    const float* wl2 = (const float*)d_in[7];
    const float* wl3 = (const float*)d_in[8];
    const float* bl2 = (const float*)d_in[9];
    const float* bl3 = (const float*)d_in[10];
    const float *g1=(const float*)d_in[11], *b1=(const float*)d_in[12], *m1=(const float*)d_in[13], *v1=(const float*)d_in[14];
    const float *g2=(const float*)d_in[15], *b2=(const float*)d_in[16], *m2=(const float*)d_in[17], *v2=(const float*)d_in[18];
    const float *g3=(const float*)d_in[19], *b3=(const float*)d_in[20], *m3=(const float*)d_in[21], *v3=(const float*)d_in[22];
    const float *g4=(const float*)d_in[23], *b4=(const float*)d_in[24], *m4=(const float*)d_in[25], *v4=(const float*)d_in[26];
    const float *g5=(const float*)d_in[27], *b5=(const float*)d_in[28], *m5=(const float*)d_in[29], *v5=(const float*)d_in[30];
    const float *g6=(const float*)d_in[31], *b6=(const float*)d_in[32], *m6=(const float*)d_in[33], *v6=(const float*)d_in[34];
    const float *g7=(const float*)d_in[35], *b7=(const float*)d_in[36], *m7=(const float*)d_in[37], *v7=(const float*)d_in[38];

    size_t off = 0;
    auto alloc = [&](size_t bytes) {
        void* p = (char*)d_ws + off;
        off += (bytes + 255) & ~(size_t)255;
        return p;
    };
    f16*   x0p  = (f16*)  alloc((size_t)BATCH * NPTS * 32 * 2);
    f16*   xcat = (f16*)  alloc((size_t)BATCH * NPTS * 512 * 2);
    float* xx   = (float*)alloc((size_t)BATCH * NPTS * 4);
    int*   idx  = (int*)  alloc((size_t)BATCH * NPTS * KNN * 4);
    float* Z    = (float*)alloc((size_t)BATCH * 512 * NPTS * 4);
    f16*   we1  = (f16*)  alloc((size_t)128 * 32 * 2);
    f16*   we2  = (f16*)  alloc((size_t)128 * 64 * 2);
    f16*   we3  = (f16*)  alloc((size_t)256 * 64 * 2);
    f16*   we4  = (f16*)  alloc((size_t)512 * 128 * 2);
    f16*   w5h  = (f16*)  alloc((size_t)1024 * 512 * 2);
    f16*   wl1h = (f16*)  alloc((size_t)512 * 2048 * 2);
    f16*   wl2h = (f16*)  alloc((size_t)256 * 512 * 2);
    f16*   wl3h = (f16*)  alloc((size_t)112 * 256 * 2);
    float* pmax = (float*)alloc((size_t)BATCH * NPTS * 64 * 4);
    float* psum = (float*)alloc((size_t)BATCH * NPTS * 64 * 4);
    f16*   p16  = (f16*)  alloc((size_t)BATCH * 2048 * 2);
    f16*   h1   = (f16*)  alloc((size_t)BATCH * 512 * 2);
    f16*   h2   = (f16*)  alloc((size_t)BATCH * 256 * 2);

    auto gdim = [](int n, int bs) { return (n + bs - 1) / bs; };

    // prep
    prep_x0_k<<<gdim(BATCH * NPTS * 32, 256), 256, 0, stream>>>(x, x0p);
    prep_edge_w_k<<<gdim(128 * 32, 256),  256, 0, stream>>>(w1, 64,  3,   32,  we1);
    prep_edge_w_k<<<gdim(128 * 64, 256),  256, 0, stream>>>(w2, 64,  64,  64,  we2);
    prep_edge_w_k<<<gdim(256 * 64, 256),  256, 0, stream>>>(w3, 128, 64,  64,  we3);
    prep_edge_w_k<<<gdim(512 * 128, 256), 256, 0, stream>>>(w4, 256, 128, 128, we4);
    prep_w_k<<<gdim(1024 * 512, 256), 256, 0, stream>>>(w5,  1024, 512,  1024, w5h);
    prep_w_k<<<gdim(512 * 2048, 256), 256, 0, stream>>>(wl1, 512,  2048, 512,  wl1h);
    prep_w_k<<<gdim(256 * 512, 256),  256, 0, stream>>>(wl2, 256,  512,  256,  wl2h);
    prep_w_k<<<gdim(112 * 256, 256),  256, 0, stream>>>(wl3, 101,  256,  112,  wl3h);

    // one EdgeConv block
    auto edge_block = [&](const f16* xin, int ldx, int coff, int Cpad,
                          const f16* Wp, int O,
                          const float* g, const float* bb, const float* mm, const float* vv,
                          int coffOut) {
        sqsum_k<<<gdim(BATCH * NPTS, 256), 256, 0, stream>>>(xin, ldx, coff, Cpad, xx);
        topk_k<<<gdim(BATCH * NPTS * 32, 256), 256, 0, stream>>>(xin, ldx, coff, Cpad, xx, idx);
        int Mtiles = (2 * O) / 16;
        int tiles = Mtiles * 64 * BATCH;
        gemm_wmma_k<<<gdim(tiles, 8), 256, 0, stream>>>(Wp, Cpad, Mtiles, xin, ldx, coff, Cpad, Z);
        edge_max_k<<<gdim(BATCH * O * NPTS, 256), 256, 0, stream>>>(Z, O, idx, g, bb, mm, vv,
                                                                    xcat, 512, coffOut);
    };

    edge_block(x0p,  32,  0,   32,  we1, 64,  g1, b1, m1, v1, 0);
    edge_block(xcat, 512, 0,   64,  we2, 64,  g2, b2, m2, v2, 64);
    edge_block(xcat, 512, 64,  64,  we3, 128, g3, b3, m3, v3, 128);
    edge_block(xcat, 512, 128, 128, we4, 256, g4, b4, m4, v4, 256);

    // conv5 + fused pooling partials
    {
        int tiles = 64 * 64 * BATCH;
        gemm5_wmma_k<<<gdim(tiles, 8), 256, 0, stream>>>(w5h, xcat, g5, b5, m5, v5, pmax, psum);
        pool_reduce_k<<<gdim(BATCH * NPTS, 256), 256, 0, stream>>>(pmax, psum, p16);
    }

    // FC head
    fc_wmma_k<<<32, 32, 0, stream>>>(p16, wl1h, 2048, 0, g6, b6, m6, v6, nullptr, h1, nullptr, 512);
    fc_wmma_k<<<16, 32, 0, stream>>>(h1,  wl2h, 512,  1, g7, b7, m7, v7, bl2,     h2, nullptr, 256);
    fc_wmma_k<<<7,  32, 0, stream>>>(h2,  wl3h, 256,  2, nullptr, nullptr, nullptr, nullptr, bl3,
                                     nullptr, (float*)d_out, 101);
}
